// Attention_49168785605080
// MI455X (gfx1250) — matrix operand order repaired and checked by reading, not compiled, not run on hardware
//
#include <hip/hip_runtime.h>
#include <hip/hip_bf16.h>
#include <math.h>

// ---- sizes from the reference ----
#define BATCH 32
#define SEQ   2048
#define HID   1024

typedef __attribute__((ext_vector_type(16))) __bf16 v16bf;
typedef __attribute__((ext_vector_type(4)))  __bf16 v4bf;
typedef __attribute__((ext_vector_type(8)))  float  v8f;
typedef __attribute__((ext_vector_type(4)))  float  f4;
typedef __attribute__((ext_vector_type(4)))  unsigned int u4;
typedef __attribute__((ext_vector_type(4)))  int v4i;

union BF16x16 { v16bf v; u4 u[2]; };

// CDNA5 async global->LDS copy (ASYNCcnt path).
// ROCm 7.2 signature: arg0 = int4 AS(1)* (global src), arg1 = int4 AS(3)* (LDS dst),
// then imm offset, imm cpol.
#if __has_builtin(__builtin_amdgcn_global_load_async_to_lds_b128)
#define HAVE_ASYNC_LDS 1
#else
#define HAVE_ASYNC_LDS 0
#endif

typedef __attribute__((address_space(1))) v4i gv4i;   // global int4
typedef __attribute__((address_space(3))) v4i lv4i;   // LDS int4

__device__ __forceinline__ void wait_async_zero() {
#if __has_builtin(__builtin_amdgcn_s_wait_asynccnt)
  __builtin_amdgcn_s_wait_asynccnt(0);
#else
  asm volatile("s_wait_asynccnt 0x0" ::: "memory");
#endif
}

// -------------------------------------------------------------------------
// 1) Convert W (fp32 row-major [o][k]) -> bf16 row-major in workspace
// -------------------------------------------------------------------------
__global__ __launch_bounds__(256) void k_convW(const float* __restrict__ W,
                                               __bf16* __restrict__ Wb) {
  const int i = (blockIdx.x * 256 + threadIdx.x) * 4;   // 1024 blocks cover 1M elems
  f4 f = *(const f4*)(W + i);
  v4bf o;
  o[0] = (__bf16)f[0]; o[1] = (__bf16)f[1];
  o[2] = (__bf16)f[2]; o[3] = (__bf16)f[3];
  *(v4bf*)(Wb + i) = o;
}

// -------------------------------------------------------------------------
// 2) hidproj[b,o] = bias[o] + sum_h hidden[b,h] * W[o,h]   (fp32, tiny GEMM)
// -------------------------------------------------------------------------
__global__ __launch_bounds__(256) void k_hidproj(const float* __restrict__ hidden,
                                                 const float* __restrict__ W,
                                                 const float* __restrict__ bias,
                                                 float* __restrict__ hp) {
  const int idx = blockIdx.x * 256 + threadIdx.x;       // 128 blocks -> 32768
  const int o = idx >> 5;
  const int b = idx & 31;
  const f4* wr = (const f4*)(W + (size_t)o * HID);
  const f4* hr = (const f4*)(hidden + (size_t)b * HID);
  float acc = bias[o];
  #pragma unroll 4
  for (int k = 0; k < HID / 4; ++k) {
    f4 w = wr[k], h = hr[k];
    acc += w[0] * h[0] + w[1] * h[1] + w[2] * h[2] + w[3] * h[3];
  }
  hp[(size_t)b * HID + o] = acc;
}

// -------------------------------------------------------------------------
// 3) zero the context region of d_out (context uses atomics)
// -------------------------------------------------------------------------
__global__ __launch_bounds__(256) void k_zero(float* __restrict__ p) {
  const int i = blockIdx.x * 256 + threadIdx.x;         // 32 blocks -> 8192 f4
  f4 z = {0.f, 0.f, 0.f, 0.f};
  ((f4*)p)[i] = z;
}

// -------------------------------------------------------------------------
// 4) scores[b,s] = sum_o v[o] * tanh( (enc @ W^T)[b,s,o] + hidproj[b,o] )
//    WMMA bf16 16x16x32, fp32 accumulate.
//    128 thr = 4 waves; wave w owns rows s0 + w*16 .. +15.
//    A fragments (full K=1024) resident in 256 VGPRs.
//    W tiles (16 x 1024 bf16, padded rows) double-buffered in dynamic LDS,
//    filled with global_load_async_to_lds_b128 overlapped with compute.
//    B fragments: 4-deep register pipeline (prefetch distance 3) so wmma
//    only waits on LDS loads issued ~3 wmma intervals earlier.
// -------------------------------------------------------------------------
#define WPAD      1032            // 1024 + 8 bf16 pad -> row stride 2064B
#define SMEM_W0   0
#define SMEM_W1   (16 * WPAD * 2)             // 33024
#define SMEM_HID  (16 * WPAD * 4)             // 66048
#define SMEM_V    (16 * WPAD * 4 + HID * 4)   // 70144
#define SMEM_TOT  (16 * WPAD * 4 + HID * 8)   // 74240 bytes

// stage W rows [nt*16, nt*16+16) into `dst` (LDS), 16 x b128 per thread
__device__ __forceinline__ void stage_tile(const __bf16* __restrict__ Wb, int nt,
                                           __bf16* dst, int tid) {
#if HAVE_ASYNC_LDS
  #pragma unroll
  for (int i = 0; i < 16; ++i) {
    const int idx = tid + i * 128;
    const int r = idx >> 7;                 // 128 u4 chunks per W row
    const int c = idx & 127;
    const void* g = (const void*)((const u4*)Wb + (size_t)(nt * 16 + r) * 128 + c);
    const void* l = (const void*)(dst + r * WPAD + c * 8);
    __builtin_amdgcn_global_load_async_to_lds_b128(
        (gv4i*)(uintptr_t)g,
        (lv4i*)(unsigned)(uintptr_t)l,   // low 32 bits of flat LDS ptr = LDS offset
        0, 0);
  }
#else
  u4 tmp[16];
  #pragma unroll
  for (int i = 0; i < 16; ++i) {
    const int idx = tid + i * 128;
    const int r = idx >> 7, c = idx & 127;
    tmp[i] = ((const u4*)Wb)[(size_t)(nt * 16 + r) * 128 + c];
  }
  #pragma unroll
  for (int i = 0; i < 16; ++i) {
    const int idx = tid + i * 128;
    const int r = idx >> 7, c = idx & 127;
    *(u4*)(dst + r * WPAD + c * 8) = tmp[i];
  }
#endif
}

__global__ __launch_bounds__(128, 1)
void k_scores(const float* __restrict__ enc,       // [B,S,H] fp32
              const __bf16* __restrict__ Wb,       // [H,H] bf16 row-major (o,k)
              const float* __restrict__ hidproj,   // [B,H]
              const float* __restrict__ vvec,      // [H]
              float* __restrict__ scores)          // [B,S]
{
  extern __shared__ char smem[];
  __bf16* ldsW0  = (__bf16*)(smem + SMEM_W0);
  __bf16* ldsW1  = (__bf16*)(smem + SMEM_W1);
  float*  ldsHid = (float*)(smem + SMEM_HID);
  float*  ldsV   = (float*)(smem + SMEM_V);

  const int tid  = threadIdx.x;
  const int wave = tid >> 5;
  const int lane = tid & 31;
  const int half = lane >> 4;          // 0: lanes 0-15, 1: lanes 16-31
  const int lrow = lane & 15;
  const int b    = blockIdx.x;         // grid (32, 32)
  const int s0   = blockIdx.y * 64;

  // stage hidproj row + v
  for (int i = tid; i < HID; i += 128) {
    ldsHid[i] = hidproj[(size_t)b * HID + i];
    ldsV[i]   = vvec[i];
  }

  // ---- preload A fragments for this wave's 16 rows, full K, into regs ----
  // A layout (16-bit, 16x32): half 0: K = {k0..k0+7, k0+16..k0+23}
  //                           half 1: K = {k0+8..k0+15, k0+24..k0+31}
  const int row = s0 + wave * 16 + lrow;
  const f4* ep = (const f4*)(enc + ((size_t)b * SEQ + row) * HID);

  v16bf a[32];
  #pragma unroll
  for (int k = 0; k < 32; ++k) {
    const int base = (k * 32 + half * 8) >> 2;   // in f4 units
    f4 f0 = ep[base], f1 = ep[base + 1];
    f4 f2 = ep[base + 4], f3 = ep[base + 5];
    v16bf t;
    t[0]  = (__bf16)f0[0]; t[1]  = (__bf16)f0[1]; t[2]  = (__bf16)f0[2]; t[3]  = (__bf16)f0[3];
    t[4]  = (__bf16)f1[0]; t[5]  = (__bf16)f1[1]; t[6]  = (__bf16)f1[2]; t[7]  = (__bf16)f1[3];
    t[8]  = (__bf16)f2[0]; t[9]  = (__bf16)f2[1]; t[10] = (__bf16)f2[2]; t[11] = (__bf16)f2[3];
    t[12] = (__bf16)f3[0]; t[13] = (__bf16)f3[1]; t[14] = (__bf16)f3[2]; t[15] = (__bf16)f3[3];
    a[k] = t;
  }

  v8f sacc = {0.f, 0.f, 0.f, 0.f, 0.f, 0.f, 0.f, 0.f};

  // prologue: async-stage tile 0 into buffer 0
  stage_tile(Wb, 0, ldsW0, tid);

  // ---- sweep all 64 n-tiles of 16 output columns ----
  for (int nt = 0; nt < 64; ++nt) {
    __bf16* cur = (nt & 1) ? ldsW1 : ldsW0;
    __bf16* nxt = (nt & 1) ? ldsW0 : ldsW1;

#if HAVE_ASYNC_LDS
    wait_async_zero();       // my async writes for `cur` have landed
#endif
    __syncthreads();         // everyone's writes landed; `nxt` free to overwrite

    if (nt < 63) stage_tile(Wb, nt + 1, nxt, tid);   // overlap copy with compute

    // B layout (32x16): n = lane%16; half 0 holds K {0..7,16..23}, half 1 K {8..15,24..31}
    const __bf16* bp = cur + lrow * WPAD + half * 8;

    // 4-deep B-fragment register pipeline (prefetch distance 3)
    BF16x16 bf[4];
    #pragma unroll
    for (int p = 0; p < 3; ++p) {
      bf[p].u[0] = *(const u4*)(bp + p * 32);
      bf[p].u[1] = *(const u4*)(bp + p * 32 + 16);
    }

    v8f c = {0.f, 0.f, 0.f, 0.f, 0.f, 0.f, 0.f, 0.f};
    #pragma unroll
    for (int k = 0; k < 32; ++k) {
      if (k < 29) {
        bf[(k + 3) & 3].u[0] = *(const u4*)(bp + (k + 3) * 32);
        bf[(k + 3) & 3].u[1] = *(const u4*)(bp + (k + 3) * 32 + 16);
      }
      c = __builtin_amdgcn_wmma_f32_16x16x32_bf16(false, a[k], false, bf[k & 3].v,
                                                  (short)0, c, false, false);
    }

    // fold v * tanh(energy + hidproj) into per-lane partial scores
    // C layout: lane holds (m = r + 8*half, n = lane%16)
    const int n = nt * 16 + lrow;
    const float hp = ldsHid[n];
    const float vn = ldsV[n];
    #pragma unroll
    for (int r = 0; r < 8; ++r)
      sacc[r] += vn * tanhf(c[r] + hp);
  }

  // reduce over the 16 n-classes: butterfly within each 16-lane half
  #pragma unroll
  for (int r = 0; r < 8; ++r) {
    float x = sacc[r];
    x += __shfl_xor(x, 1, 32);
    x += __shfl_xor(x, 2, 32);
    x += __shfl_xor(x, 4, 32);
    x += __shfl_xor(x, 8, 32);
    sacc[r] = x;
  }
  if (lrow == 0) {
    #pragma unroll
    for (int r = 0; r < 8; ++r) {
      const int m = r + 8 * half;
      scores[(size_t)b * SEQ + s0 + wave * 16 + m] = sacc[r];
    }
  }
}

// -------------------------------------------------------------------------
// 5) softmax over S per batch row: one 256-thread block per b
// -------------------------------------------------------------------------
__global__ __launch_bounds__(256) void k_softmax(const float* __restrict__ scores,
                                                 float* __restrict__ attn) {
  __shared__ float sred[8];
  const int b = blockIdx.x, tid = threadIdx.x;
  const float* src = scores + (size_t)b * SEQ;

  float vals[8];
  float m = -3.4e38f;
  #pragma unroll
  for (int i = 0; i < 8; ++i) { vals[i] = src[tid + 256 * i]; m = fmaxf(m, vals[i]); }
  #pragma unroll
  for (int off = 16; off; off >>= 1) m = fmaxf(m, __shfl_xor(m, off, 32));
  if ((tid & 31) == 0) sred[tid >> 5] = m;
  __syncthreads();
  float bm = sred[0];
  #pragma unroll
  for (int w = 1; w < 8; ++w) bm = fmaxf(bm, sred[w]);
  __syncthreads();

  float s = 0.f;
  #pragma unroll
  for (int i = 0; i < 8; ++i) { vals[i] = __expf(vals[i] - bm); s += vals[i]; }
  #pragma unroll
  for (int off = 16; off; off >>= 1) s += __shfl_xor(s, off, 32);
  if ((tid & 31) == 0) sred[tid >> 5] = s;
  __syncthreads();
  float bs = 0.f;
  #pragma unroll
  for (int w = 0; w < 8; ++w) bs += sred[w];
  const float inv = 1.0f / bs;
  #pragma unroll
  for (int i = 0; i < 8; ++i) attn[(size_t)b * SEQ + tid + 256 * i] = vals[i] * inv;
}

// -------------------------------------------------------------------------
// 6) context[b,h] = sum_s attn[b,s] * enc[b,s,h]   (memory bound, 256 MB read)
// -------------------------------------------------------------------------
__global__ __launch_bounds__(256) void k_context(const float* __restrict__ enc,
                                                 const float* __restrict__ attn,
                                                 float* __restrict__ ctx) {
  const int b  = blockIdx.x;
  const int h  = blockIdx.y * 256 + threadIdx.x;
  const int s0 = blockIdx.z * 256;
  const float* e = enc + ((size_t)b * SEQ + s0) * HID + h;
  const float* a = attn + (size_t)b * SEQ + s0;
  float acc = 0.f;
  #pragma unroll 8
  for (int s = 0; s < 256; ++s)
    acc += a[s] * e[(size_t)s * HID];
  atomicAdd(&ctx[(size_t)b * HID + h], acc);
}

// -------------------------------------------------------------------------
// launch
// -------------------------------------------------------------------------
extern "C" void kernel_launch(void* const* d_in, const int* in_sizes, int n_in,
                              void* d_out, int out_size, void* d_ws, size_t ws_size,
                              hipStream_t stream) {
  (void)in_sizes; (void)n_in; (void)out_size; (void)ws_size;
  const float* hidden = (const float*)d_in[0];   // [32,1024]
  const float* enc    = (const float*)d_in[1];   // [32,2048,1024]
  const float* W      = (const float*)d_in[2];   // [1024,1024]
  const float* bias   = (const float*)d_in[3];   // [1024]
  const float* vvec   = (const float*)d_in[4];   // [1024]

  float* out  = (float*)d_out;
  float* ctx  = out;                   // [32,1024]
  float* attn = out + BATCH * HID;     // [32,2048]

  char* ws = (char*)d_ws;
  __bf16* Wb      = (__bf16*)ws;                                     // 2 MB
  float*  hidproj = (float*)(ws + (size_t)2 * 1024 * 1024);          // 128 KB
  float*  scores  = (float*)(ws + (size_t)2 * 1024 * 1024 + 131072); // 256 KB

  k_convW   <<<1024, 256, 0, stream>>>(W, Wb);
  k_hidproj <<<128,  256, 0, stream>>>(hidden, W, bias, hidproj);
  k_zero    <<<32,   256, 0, stream>>>(ctx);

  dim3 gs(BATCH, SEQ / 64);
  k_scores  <<<gs, 128, SMEM_TOT, stream>>>(enc, Wb, hidproj, vvec, scores);

  k_softmax <<<BATCH, 256, 0, stream>>>(scores, attn);

  dim3 gc(BATCH, HID / 256, SEQ / 256);
  k_context <<<gc, 256, 0, stream>>>(enc, attn, ctx);
}
